// DatastoreReaderLayer_69784628626063
// MI455X (gfx1250) — matrix operand order repaired and checked by reading, not compiled, not run on hardware
//
#include <hip/hip_runtime.h>

#define SEQ     256
#define BATCH   4
#define DMODEL  512
#define NKEYS   32768
#define NSPLIT  8
#define KSLEN   (NKEYS / NSPLIT)

typedef __bf16 bf16_t;
typedef __attribute__((ext_vector_type(16))) __bf16 v16bf;
typedef __attribute__((ext_vector_type(8)))  float  v8f;

static __device__ __forceinline__ v8f wmma_bf16(v16bf a, v16bf b, v8f c) {
  // (neg_a, A, neg_b, B, c_mod, C, reuse_a, reuse_b)
  return __builtin_amdgcn_wmma_f32_16x16x32_bf16(false, a, false, b, (short)0, c,
                                                 false, false);
}

static __device__ __forceinline__ v8f vzero8() {
  v8f z;
#pragma unroll
  for (int i = 0; i < 8; ++i) z[i] = 0.0f;
  return z;
}

// ---- gfx1250 async global->LDS copy (ASYNCcnt-tracked) ---------------------
// VFLAT async encoding: VDST = LDS byte address VGPR, VADDR = 64-bit global
// address (CDNA5 ISA).
static __device__ __forceinline__ void async_ld_b128(const void* g, void* l) {
  const uint32_t lds_off = (uint32_t)(uintptr_t)l;
  const uint64_t gaddr   = (uint64_t)(uintptr_t)g;
  asm volatile("global_load_async_to_lds_b128 %0, %1, off"
               :: "v"(lds_off), "v"(gaddr)
               : "memory");
}

static __device__ __forceinline__ void async_wait0() {
#if __has_builtin(__builtin_amdgcn_s_wait_asynccnt)
  __builtin_amdgcn_s_wait_asynccnt(0);
#else
  asm volatile("s_wait_asynccnt 0x0" ::: "memory");
#endif
}

// A fragment (16x32 bf16, M x K) from an f32 row pointer.
// Lanes 0-15 -> M=lane, K = 0..15; lanes 16-31 same M, K = 16..31.
static __device__ __forceinline__ v16bf load_a_from_f32(const float* __restrict__ rowp,
                                                        int kd, int half) {
  v16bf a;
  const float* p0 = rowp + kd + 16 * half;
  const float* p1 = rowp + kd + 16 * half + 8;
#pragma unroll
  for (int i = 0; i < 8; ++i) {
    a[i]     = (__bf16)p0[i];
    a[8 + i] = (__bf16)p1[i];
  }
  return a;
}

// B fragment (32x16 bf16, K x N) from an f32 row pointer: 16 contiguous K values.
static __device__ __forceinline__ v16bf load_b_from_f32(const float* __restrict__ rowp,
                                                        int off) {
  v16bf b;
#pragma unroll
  for (int i = 0; i < 16; ++i) b[i] = (__bf16)rowp[off + i];
  return b;
}

// A fragment from LDS bf16 row (two 16B chunks per lane).
static __device__ __forceinline__ v16bf load_a_from_lds(const bf16_t* rowp, int kd, int half) {
  union { v16bf v; uint4 u[2]; } t;
  t.u[0] = *(const uint4*)(rowp + kd + 16 * half);
  t.u[1] = *(const uint4*)(rowp + kd + 16 * half + 8);
  return t.v;
}

// A fragment from f32 LDS scratch (P tile, 16x32 row-major floats).
static __device__ __forceinline__ v16bf load_a_from_f32lds(const float* rowp, int half) {
  v16bf a;
  const float* p0 = rowp + 16 * half;
  const float* p1 = rowp + 16 * half + 8;
#pragma unroll
  for (int i = 0; i < 8; ++i) {
    a[i]     = (__bf16)p0[i];
    a[8 + i] = (__bf16)p1[i];
  }
  return a;
}

// ---------------------------------------------------------------------------
// Kernel 1: out[m, j] = (sum_d in[m,d] * W[j,d] + bias[j]) * scale  -> bf16
// in_seqbatch: row m maps to input element ((m%SEQ)*BATCH + m/SEQ) (for q [S,B,D]).
// out_transposed: write out[j*M + m] (for v -> vT[D, N]); else out[m*DMODEL + j].
// One wave per 16x16 output tile. 8 waves / block.
// ---------------------------------------------------------------------------
__global__ __launch_bounds__(256) void proj_wmma(
    const float* __restrict__ in, const float* __restrict__ W,
    const float* __restrict__ bias, bf16_t* __restrict__ out,
    int M, float scale, int in_seqbatch, int out_transposed) {
  const int wave = blockIdx.x * 8 + (threadIdx.x >> 5);
  const int lane = threadIdx.x & 31;
  const int half = lane >> 4;
  const int ln   = lane & 15;

  const int colTiles = DMODEL / 16;           // 32
  const int mt = wave / colTiles;
  const int jt = wave % colTiles;
  const int m0 = mt * 16, j0 = jt * 16;

  const int arow = m0 + ln;
  size_t abase;
  if (in_seqbatch)
    abase = (size_t)(((arow & (SEQ - 1)) * BATCH) + (arow >> 8)) * DMODEL;
  else
    abase = (size_t)arow * DMODEL;
  const float* arp = in + abase;
  const float* brp = W + (size_t)(j0 + ln) * DMODEL;

  v8f acc = vzero8();
#pragma unroll 4
  for (int kd = 0; kd < DMODEL; kd += 32) {
    v16bf a = load_a_from_f32(arp, kd, half);
    v16bf b = load_b_from_f32(brp, kd + 16 * half);
    acc = wmma_bf16(a, b, acc);
  }

  const float bv = bias[j0 + ln];
  const int col = j0 + ln;
#pragma unroll
  for (int r = 0; r < 8; ++r) {
    const int row = m0 + r + 8 * half;
    const float v = (acc[r] + bv) * scale;
    if (out_transposed)
      out[(size_t)col * M + row] = (__bf16)v;
    else
      out[(size_t)row * DMODEL + col] = (__bf16)v;
  }
}

// ---------------------------------------------------------------------------
// Kernel 2: flash attention over the datastore (partial per key-split).
// Block = 128 threads = 4 waves; wave w owns d-slice [w*128, w*128+128).
// K chunk (32 keys x 512 bf16 = 32KB) staged in LDS once per block via
// async global->LDS copies, shared by all 4 waves.
// blockIdx.x = qtile * NSPLIT + ksplit.
// ---------------------------------------------------------------------------
__global__ __launch_bounds__(128) void attn_flash(
    const bf16_t* __restrict__ qb, const bf16_t* __restrict__ kb,
    const bf16_t* __restrict__ vt,
    float* __restrict__ pO, float* __restrict__ pM, float* __restrict__ pL) {
  __shared__ bf16_t qtile[16 * DMODEL];      // 16 KB
  __shared__ bf16_t kbuf[32 * DMODEL];       // 32 KB (staged K chunk)
  __shared__ float  pscr[4][16 * 32];        // 8 KB (per-wave P transpose scratch)

  const int qt = blockIdx.x / NSPLIT;
  const int ks = blockIdx.x % NSPLIT;
  const int tid  = threadIdx.x;
  const int wv   = tid >> 5;
  const int lane = tid & 31;
  const int half = lane >> 4;
  const int ln   = lane & 15;

  // Cooperative Q tile load (16 rows x 512 bf16 = 1024 uint4).
  {
    const uint4* src = (const uint4*)(qb + (size_t)qt * 16 * DMODEL);
    uint4* dst = (uint4*)qtile;
    for (int i = tid; i < 1024; i += 128) dst[i] = src[i];
  }
  __syncthreads();

  const int d0 = wv * 128;
  v8f O[8];
#pragma unroll
  for (int t = 0; t < 8; ++t) O[t] = vzero8();
  float mrow[8], lrow[8];
#pragma unroll
  for (int r = 0; r < 8; ++r) { mrow[r] = -3.0e38f; lrow[r] = 0.0f; }

  const int k0 = ks * KSLEN;
  const bf16_t* qrow = qtile + (size_t)ln * DMODEL;
  float* sc = &pscr[wv][0];

  for (int kc = 0; kc < KSLEN; kc += 32) {
    const int kbase = k0 + kc;

    // Stage 32 K rows (32 KB) into LDS: 16 async b128 per thread.
    {
      const char* gsrc = (const char*)(kb + (size_t)kbase * DMODEL);
      char* ldst = (char*)kbuf;
#pragma unroll
      for (int i = 0; i < 16; ++i) {
        const int off = (i * 128 + tid) * 16;
        async_ld_b128(gsrc + off, ldst + off);
      }
      __builtin_prefetch((const void*)(gsrc + 32 * DMODEL * 2), 0, 1);
      async_wait0();
    }
    __syncthreads();

    const bf16_t* krow0 = kbuf + (size_t)ln * DMODEL;
    const bf16_t* krow1 = kbuf + (size_t)(16 + ln) * DMODEL;

    v8f S0 = vzero8(), S1 = vzero8();
#pragma unroll 4
    for (int kd = 0; kd < DMODEL; kd += 32) {
      v16bf a   = load_a_from_lds(qrow, kd, half);
      v16bf bk0 = *(const v16bf*)(krow0 + kd + 16 * half);
      S0 = wmma_bf16(a, bk0, S0);
      v16bf bk1 = *(const v16bf*)(krow1 + kd + 16 * half);
      S1 = wmma_bf16(a, bk1, S1);
    }

    // Online softmax stats. Row m = r + 8*half lives across the 16 lanes of
    // this half -> butterfly over lane masks 1,2,4,8.
    float corr[8];
#pragma unroll
    for (int r = 0; r < 8; ++r) {
      float mx = fmaxf(S0[r], S1[r]);
#pragma unroll
      for (int sh = 1; sh < 16; sh <<= 1) mx = fmaxf(mx, __shfl_xor(mx, sh, 32));
      const float mn = fmaxf(mrow[r], mx);
      corr[r] = __expf(mrow[r] - mn);
      mrow[r] = mn;
      const float p0 = __expf(S0[r] - mn);
      const float p1 = __expf(S1[r] - mn);
      S0[r] = p0; S1[r] = p1;
      float rs = p0 + p1;
#pragma unroll
      for (int sh = 1; sh < 16; sh <<= 1) rs += __shfl_xor(rs, sh, 32);
      lrow[r] = lrow[r] * corr[r] + rs;
    }

    // Rescale accumulators.
#pragma unroll
    for (int t = 0; t < 8; ++t)
#pragma unroll
      for (int r = 0; r < 8; ++r) O[t][r] *= corr[r];

    // P (C layout) -> LDS row-major 16x32 -> reload as A fragment.
#pragma unroll
    for (int r = 0; r < 8; ++r) {
      sc[(r + 8 * half) * 32 + ln]      = S0[r];
      sc[(r + 8 * half) * 32 + 16 + ln] = S1[r];
    }
    asm volatile("s_wait_dscnt 0x0" ::: "memory");
    v16bf pa = load_a_from_f32lds(sc + ln * 32, half);

    // O += P x V   (V transposed: vt[d, key] contiguous in key)
#pragma unroll
    for (int t = 0; t < 8; ++t) {
      const bf16_t* vrow =
          vt + (size_t)(d0 + t * 16 + ln) * NKEYS + kbase + 16 * half;
      v16bf bV = *(const v16bf*)vrow;
      O[t] = wmma_bf16(pa, bV, O[t]);
    }

    __syncthreads();   // all reads of kbuf done before next chunk's writes
  }

  // Write partials.
  const int pb = (qt * NSPLIT + ks) * 16;
#pragma unroll
  for (int t = 0; t < 8; ++t)
#pragma unroll
    for (int r = 0; r < 8; ++r)
      pO[(size_t)(pb + r + 8 * half) * DMODEL + d0 + t * 16 + ln] = O[t][r];
  if (wv == 0 && ln == 0) {
#pragma unroll
    for (int r = 0; r < 8; ++r) {
      pM[pb + r + 8 * half] = mrow[r];
      pL[pb + r + 8 * half] = lrow[r];
    }
  }
}

// ---------------------------------------------------------------------------
// Kernel 3: combine key-split partials -> attn [1024, 512] f32.
// ---------------------------------------------------------------------------
__global__ __launch_bounds__(256) void attn_reduce(
    const float* __restrict__ pO, const float* __restrict__ pM,
    const float* __restrict__ pL, float* __restrict__ attn) {
  const int qt = blockIdx.x;
  const int tid = threadIdx.x;
  __shared__ float wexp[NSPLIT][16];
  __shared__ float Lr[16];

  if (tid < 16) {
    const int row = tid;
    float M = -3.0e38f;
#pragma unroll
    for (int s = 0; s < NSPLIT; ++s)
      M = fmaxf(M, pM[(qt * NSPLIT + s) * 16 + row]);
    float L = 0.0f;
#pragma unroll
    for (int s = 0; s < NSPLIT; ++s) {
      const float e = __expf(pM[(qt * NSPLIT + s) * 16 + row] - M);
      wexp[s][row] = e;
      L += e * pL[(qt * NSPLIT + s) * 16 + row];
    }
    Lr[row] = (L > 0.0f) ? L : 1.0f;
  }
  __syncthreads();

  for (int idx = tid; idx < 16 * DMODEL; idx += 256) {
    const int row = idx >> 9;
    const int d = idx & (DMODEL - 1);
    float acc = 0.0f;
#pragma unroll
    for (int s = 0; s < NSPLIT; ++s)
      acc += wexp[s][row] * pO[(size_t)((qt * NSPLIT + s) * 16 + row) * DMODEL + d];
    attn[(size_t)(qt * 16 + row) * DMODEL + d] = acc / Lr[row];
  }
}

// ---------------------------------------------------------------------------
// Kernel 4: gated residual fusion. One block per query row (b*SEQ+s).
// ---------------------------------------------------------------------------
__global__ __launch_bounds__(256) void gate_fuse(
    const float* __restrict__ attn, const float* __restrict__ prev,
    const float* __restrict__ Wg1, const float* __restrict__ bg1,
    const float* __restrict__ Wg2, const float* __restrict__ bg2,
    float* __restrict__ out) {
  const int row = blockIdx.x;              // b*SEQ + s
  const int s = row & (SEQ - 1);
  const int b = row >> 8;
  const size_t pidx = ((size_t)s * BATCH + b) * DMODEL;  // [S,B,D] index
  const int tid = threadIdx.x;

  __shared__ float cat[2 * DMODEL];
  __shared__ float hred[256];

  for (int i = tid; i < DMODEL; i += 256) {
    cat[i]          = attn[(size_t)row * DMODEL + i];
    cat[DMODEL + i] = prev[pidx + i];
  }
  __syncthreads();

  float part = 0.0f;
#pragma unroll
  for (int jj = 0; jj < 2; ++jj) {
    const int j = tid + jj * 256;
    const float* wr = Wg1 + (size_t)j * (2 * DMODEL);
    float acc = bg1[j];
    for (int t = 0; t < 2 * DMODEL; ++t) acc += cat[t] * wr[t];
    const float h = fmaxf(acc, 0.0f);
    part += h * Wg2[j];
  }
  hred[tid] = part;
  __syncthreads();
  for (int st = 128; st > 0; st >>= 1) {
    if (tid < st) hred[tid] += hred[tid + st];
    __syncthreads();
  }
  const float sig = 1.0f / (1.0f + __expf(-(hred[0] + bg2[0])));

  for (int i = tid; i < DMODEL; i += 256) {
    const float a = cat[i];
    const float p = cat[DMODEL + i];
    out[pidx + i] = a * sig + p * (1.0f - sig);
  }
}

// ---------------------------------------------------------------------------
extern "C" void kernel_launch(void* const* d_in, const int* in_sizes, int n_in,
                              void* d_out, int out_size, void* d_ws, size_t ws_size,
                              hipStream_t stream) {
  const float* q    = (const float*)d_in[0];
  const float* prev = (const float*)d_in[1];
  const float* Wq   = (const float*)d_in[2];
  const float* bq   = (const float*)d_in[3];
  const float* Wk   = (const float*)d_in[4];
  const float* bk   = (const float*)d_in[5];
  const float* Wv   = (const float*)d_in[6];
  const float* bv   = (const float*)d_in[7];
  const float* Wg1  = (const float*)d_in[8];
  const float* bg1  = (const float*)d_in[9];
  const float* Wg2  = (const float*)d_in[10];
  const float* bg2  = (const float*)d_in[11];
  const float* dsk  = (const float*)d_in[12];
  const float* dsv  = (const float*)d_in[13];
  float* out = (float*)d_out;

  // Workspace carving.
  char* w = (char*)d_ws;
  bf16_t* qb16 = (bf16_t*)w;                                     // 1 MB
  bf16_t* kb16 = (bf16_t*)(w + (1u << 20));                      // 32 MB
  bf16_t* vt16 = (bf16_t*)(w + (1u << 20) + (32u << 20));        // 32 MB
  float*  pO   = (float*)(w + (1u << 20) + (64u << 20));         // 16 MB
  float*  pM   = (float*)(w + (1u << 20) + (80u << 20));         // 32 KB
  float*  pL   = (float*)(w + (1u << 20) + (80u << 20) + (1u << 16));
  float*  attn = (float*)(w + (1u << 20) + (80u << 20) + (2u << 16)); // 2 MB

  const float qscale = 0.08838834764831845f;  // D^-0.5 / TEMP

  // Projections (WMMA bf16).
  proj_wmma<<<(64 * 32) / 8, 256, 0, stream>>>(q, Wq, bq, qb16, SEQ * BATCH,
                                               qscale, 1, 0);
  proj_wmma<<<(2048 * 32) / 8, 256, 0, stream>>>(dsk, Wk, bk, kb16, NKEYS,
                                                 1.0f, 0, 0);
  proj_wmma<<<(2048 * 32) / 8, 256, 0, stream>>>(dsv, Wv, bv, vt16, NKEYS,
                                                 1.0f, 0, 1);
  // Flash attention partials + reduce.
  attn_flash<<<64 * NSPLIT, 128, 0, stream>>>(qb16, kb16, vt16, pO, pM, pL);
  attn_reduce<<<64, 256, 0, stream>>>(pO, pM, pL, attn);
  // Gated residual fusion.
  gate_fuse<<<SEQ * BATCH, 256, 0, stream>>>(attn, prev, Wg1, bg1, Wg2, bg2, out);
}
